// StableLMAttention_6502580486538
// MI455X (gfx1250) — matrix operand order repaired and checked by reading, not compiled, not run on hardware
//
#include <hip/hip_runtime.h>
#include <hip/hip_bf16.h>

// ---------------------------------------------------------------- constants
typedef __bf16 bf16_t;
typedef __attribute__((ext_vector_type(16))) __bf16 v16bf;
typedef __attribute__((ext_vector_type(8)))  float  v8f;
typedef __attribute__((ext_vector_type(4)))  unsigned int u32x4;
typedef __attribute__((ext_vector_type(8)))  int i32x8;
typedef __attribute__((ext_vector_type(4)))  int i32x4;

constexpr int B_   = 2;
constexpr int S_   = 2048;
constexpr int M_   = 2560;
constexpr int NQ_  = 32;
constexpr int NKV_ = 8;
constexpr int HD_  = 80;
constexpr int HDP_ = 96;              // head dim padded to 3x32 for WMMA K-steps
constexpr int NT_  = B_ * S_;         // 4096 tokens
constexpr int QCOLS = NQ_ * HD_;      // 2560
constexpr int KVCOLS = NKV_ * HD_;    // 640
constexpr float INV_NORM_ = 0.11180339887498949f;  // 1/sqrt(80)
constexpr float NEG_ = -1.0e9f;

#if defined(__has_builtin)
#if __has_builtin(__builtin_amdgcn_tensor_load_to_lds) && \
    __has_builtin(__builtin_amdgcn_s_wait_tensorcnt)
#define USE_TDM 1
#endif
#endif
#ifndef USE_TDM
#define USE_TDM 0
#endif

// ---------------------------------------------------------------- helpers
__device__ inline v8f v8f_zero() {
  v8f z;
#pragma unroll
  for (int i = 0; i < 8; ++i) z[i] = 0.0f;
  return z;
}

union FragU {
  v16bf v;
  uint4 q[2];
};

// A fragment (16x32, 16-bit): lane = row (lane&15), half = lane>>4.
// elements 0..7  <- k_base + 8*half + [0..7]   (16 contiguous bytes)
// elements 8..15 <- k_base + 16 + 8*half + [0..7]
__device__ inline v16bf frag_a_row(const bf16_t* tile, int ld, int kbase, int lane) {
  const int r = lane & 15, h = lane >> 4;
  const bf16_t* p = tile + (size_t)r * ld + kbase + 8 * h;
  FragU f;
  f.q[0] = *(const uint4*)(p);
  f.q[1] = *(const uint4*)(p + 16);
  return f.v;
}

// B fragment (32x16, 16-bit), sourced from a transposed (row = n) buffer:
// lane n = lane&15, khalf = lane>>4; elements 0..7 <- k_base + 8*khalf + [0..7], 8..15 <- k_base + 16 + 8*khalf + [0..7]
__device__ inline v16bf frag_b_rowT(const bf16_t* tileT, int ld, int kbase, int lane) {
  const int n = lane & 15, kh = lane >> 4;
  const bf16_t* p = tileT + (size_t)n * ld + kbase + 8 * kh;
  FragU f;
  f.q[0] = *(const uint4*)(p);
  f.q[1] = *(const uint4*)(p + 16);
  return f.v;
}

__device__ inline v8f wmma_bf16(v16bf a, v16bf b, v8f c) {
  return __builtin_amdgcn_wmma_f32_16x16x32_bf16(
      /*neg_a=*/false, a, /*neg_b=*/false, b,
      /*c_mod=*/(short)0, c, /*reuse_a=*/false, /*reuse_b=*/false);
}

// C/D layout: lane n = lane&15, rows m = i + 8*(lane>>4), i = 0..7
__device__ inline void store_c_bf16(bf16_t* tile, int ld, v8f c, int lane) {
  const int n = lane & 15, h = lane >> 4;
#pragma unroll
  for (int i = 0; i < 8; ++i) tile[(size_t)(i + 8 * h) * ld + n] = (bf16_t)c[i];
}
__device__ inline void store_c_f32(float* tile, int ld, v8f c, int lane) {
  const int n = lane & 15, h = lane >> 4;
#pragma unroll
  for (int i = 0; i < 8; ++i) tile[(size_t)(i + 8 * h) * ld + n] = c[i];
}

#if USE_TDM
// Tensor Data Mover: DMA a 2D tile (tile_h rows x tile_w bf16 elems, row stride
// in elements) from global memory into a packed LDS block. D# per
// CDNA5 ISA: group0 = {count/flags, lds_addr,
// global_addr, type=2}; group1 = {mask/data_size, tensor dims, tile dims,
// dim0 stride}. One instruction moves the whole tile; tracked by TENSORcnt.
__device__ inline void tdm_load_2d(const bf16_t* gsrc, void* lds_dst,
                                   int tile_w, int tile_h, int stride_elems) {
  unsigned long long ga = (unsigned long long)(size_t)gsrc;
  unsigned int laddr = (unsigned int)(size_t)lds_dst;  // low 32 bits = LDS byte addr
  u32x4 g0;
  g0[0] = 1u;                                         // count=1 (valid user D#)
  g0[1] = laddr;                                      // lds_addr
  g0[2] = (unsigned int)(ga & 0xffffffffu);           // global_addr[31:0]
  g0[3] = (unsigned int)((ga >> 32) & 0x1ffffffu) | (2u << 30);  // ga[56:32] | type=2
  const unsigned int td0 = 0x40000000u;               // huge tensor dims: no OOB clip
  const unsigned int td1 = 0x40000000u;
  const unsigned long long s0 = (unsigned long long)(unsigned int)stride_elems;
  i32x8 g1;
  g1[0] = (int)(1u << 16);                            // wg_mask=0, data_size=1 (2B)
  g1[1] = (int)((td0 & 0xffffu) << 16);               // tensor_dim0[15:0]
  g1[2] = (int)((td0 >> 16) | ((td1 & 0xffffu) << 16));
  g1[3] = (int)((td1 >> 16) | ((unsigned int)tile_w << 16));  // tile_dim0
  g1[4] = (int)(unsigned int)tile_h;                  // tile_dim1 (tile_dim2=0)
  g1[5] = (int)(unsigned int)(s0 & 0xffffffffu);      // tensor_dim0_stride[31:0]
  g1[6] = (int)(unsigned int)((s0 >> 32) & 0xffffu);  // stride[47:32] | dim1_stride lo = 0
  g1[7] = 0;
  i32x4 z4;
  z4[0] = 0; z4[1] = 0; z4[2] = 0; z4[3] = 0;
  i32x8 z8;
#pragma unroll
  for (int i = 0; i < 8; ++i) z8[i] = 0;
  // 6-arg form (clang-23 / therock-10.0 headers): groups 0..3 + aux + cpol
  __builtin_amdgcn_tensor_load_to_lds(g0, g1, z4, z4, z8, 0);
}
#endif

// ---------------------------------------------------------------- conversion
__global__ void k_convert(const float* __restrict__ src, bf16_t* __restrict__ dst, int n) {
  int i = blockIdx.x * blockDim.x + threadIdx.x;
  if (i < n) dst[i] = (bf16_t)src[i];
}

// dst[c*R + r] = src[r*C + c]  (W [R,C] f32 -> Wt [C,R] bf16)
__global__ void k_convert_t(const float* __restrict__ src, bf16_t* __restrict__ dst,
                            int R, int C) {
  int i = blockIdx.x * blockDim.x + threadIdx.x;
  if (i >= R * C) return;
  int r = i / C, c = i - r * C;
  dst[(size_t)c * R + r] = (bf16_t)src[i];
}

// ---------------------------------------------------------------- GEMM
// C[rows, N] = A[rows, K] * Bt[N, K]^T ; A,Bt bf16; acc f32.
// 256 threads = 8 waves (4 row-waves x 2 col-waves); block tile 128x128;
// each wave owns a 32x64 register tile = 2x4 WMMA accumulators (8 WMMA/K-step).
// K staged 32-deep, double-buffered LDS; tiles DMA'd by the TDM when available.
template <bool OUTF32>
__global__ void __launch_bounds__(256)
k_gemm(const bf16_t* __restrict__ A, const bf16_t* __restrict__ Bt,
       void* __restrict__ Cv, int K, int ldA, int ldBt, int ldC) {
  __shared__ alignas(16) bf16_t As[2][128 * 32];
  __shared__ alignas(16) bf16_t Bs[2][128 * 32];

  const int tid = threadIdx.x;
  const int lane = tid & 31;
  const int wave = tid >> 5;
  const int wm = wave & 3;       // 0..3 (row waves)
  const int wn = wave >> 2;      // 0..1 (col waves)
  const int row0 = blockIdx.x * 128;
  const int col0 = blockIdx.y * 128;

  v8f acc[2][4];
#pragma unroll
  for (int i = 0; i < 2; ++i)
#pragma unroll
    for (int j = 0; j < 4; ++j) acc[i][j] = v8f_zero();

  auto stage = [&](int buf, int k) {
#if USE_TDM
    if (wave == 0) {
      tdm_load_2d(A + (size_t)row0 * ldA + k, &As[buf][0], 32, 128, ldA);
      tdm_load_2d(Bt + (size_t)col0 * ldBt + k, &Bs[buf][0], 32, 128, ldBt);
    }
#else
    const int r = tid >> 1;
    const int c = (tid & 1) * 16;
    *(uint4*)&As[buf][r * 32 + c]     = *(const uint4*)&A[(size_t)(row0 + r) * ldA + k + c];
    *(uint4*)&As[buf][r * 32 + c + 8] = *(const uint4*)&A[(size_t)(row0 + r) * ldA + k + c + 8];
    *(uint4*)&Bs[buf][r * 32 + c]     = *(const uint4*)&Bt[(size_t)(col0 + r) * ldBt + k + c];
    *(uint4*)&Bs[buf][r * 32 + c + 8] = *(const uint4*)&Bt[(size_t)(col0 + r) * ldBt + k + c + 8];
#endif
  };

  stage(0, 0);
  for (int k = 0; k < K; k += 32) {
    const int cur = (k >> 5) & 1;
    const bool hasnext = (k + 32) < K;
    if (hasnext) stage(cur ^ 1, k + 32);
#if USE_TDM
    if (wave == 0) {
      // TDM completes in order per wave: <=2 outstanding means the current
      // buffer's two tile loads have landed. Last step waits for everything.
      if (hasnext) __builtin_amdgcn_s_wait_tensorcnt((short)2);
      else         __builtin_amdgcn_s_wait_tensorcnt((short)0);
    }
#endif
    __syncthreads();

    const bf16_t* as = &As[cur][0];
    const bf16_t* bs = &Bs[cur][0];
    v16bf a0 = frag_a_row(as + (wm * 32 + 0) * 32, 32, 0, lane);
    v16bf a1 = frag_a_row(as + (wm * 32 + 16) * 32, 32, 0, lane);
#pragma unroll
    for (int j = 0; j < 4; ++j) {
      v16bf bfr = frag_b_rowT(bs + (wn * 64 + j * 16) * 32, 32, 0, lane);
      acc[0][j] = wmma_bf16(a0, bfr, acc[0][j]);
      acc[1][j] = wmma_bf16(a1, bfr, acc[1][j]);
    }
    __syncthreads();
  }

#pragma unroll
  for (int sm = 0; sm < 2; ++sm) {
#pragma unroll
    for (int j = 0; j < 4; ++j) {
      const int crow = row0 + wm * 32 + sm * 16;
      const int ccol = col0 + wn * 64 + j * 16;
      if constexpr (OUTF32)
        store_c_f32((float*)Cv + (size_t)crow * ldC + ccol, ldC, acc[sm][j], lane);
      else
        store_c_bf16((bf16_t*)Cv + (size_t)crow * ldC + ccol, ldC, acc[sm][j], lane);
    }
  }
}

// ---------------------------------------------------------------- RoPE + pad
// src: [NT, nheads*80] bf16 ; dst: [NT, nheads, 96] bf16 (dims 80..95 zero)
__global__ void k_rope_pack(const bf16_t* __restrict__ src, bf16_t* __restrict__ dst,
                            int nheads, int total) {
  int t = blockIdx.x * blockDim.x + threadIdx.x;
  if (t >= total) return;
  int d = t % HDP_;
  int rest = t / HDP_;
  int head = rest % nheads;
  int tok = rest / nheads;
  int s = tok & (S_ - 1);

  float out;
  if (d >= HD_) {
    out = 0.0f;
  } else {
    const size_t base = (size_t)tok * nheads * HD_ + (size_t)head * HD_;
    float x = (float)src[base + d];
    if (d < 20) {
      int i = d % 10;
      float fr = (float)i * 0.1f;                 // 2i/20
      float inv = __powf(10000.0f, -fr);
      float ang = (float)s * inv;
      float sn, cs;
      __sincosf(ang, &sn, &cs);
      float other = (d < 10) ? -(float)src[base + d + 10]
                             :  (float)src[base + d - 10];
      out = x * cs + other * sn;
    } else {
      out = x;
    }
  }
  dst[(size_t)tok * nheads * HDP_ + (size_t)head * HDP_ + d] = (bf16_t)out;
}

// ---------------------------------------------------------------- attention
// 1 wave per (b, head, 16-row q tile). Flash-style online softmax in fp32.
// Qp: [NT, NQ, 96]  Kp: [NT, NKV, 96]  Vr: [NT, NKV*80]  out: [NT, NQ*80] bf16
__global__ void __launch_bounds__(32)
k_attn(const bf16_t* __restrict__ Qp, const bf16_t* __restrict__ Kp,
       const bf16_t* __restrict__ Vr, bf16_t* __restrict__ attn_out) {
  constexpr int QT = S_ / 16;  // 128 q-tiles per (b, head)
  const int bid = blockIdx.x;
  const int qt = bid % QT;
  const int u  = (bid / QT) % NQ_;
  const int b  = bid / (QT * NQ_);
  const int kvh = u >> 2;  // GROUPS = 4
  const int lane = threadIdx.x;
  const int n = lane & 15, half = lane >> 4;

  __shared__ alignas(16) float  Sls[16][32];
  __shared__ alignas(16) bf16_t Pls[16][32];
  __shared__ alignas(16) bf16_t Vls[80][32];
  __shared__ float mrow[16], lrow[16], crow[16];

  const size_t qrow0 = (size_t)(b * S_ + qt * 16);
  const bf16_t* qbase = Qp + qrow0 * (NQ_ * HDP_) + (size_t)u * HDP_;
  const int ldQ = NQ_ * HDP_;
  v16bf qf0 = frag_a_row(qbase, ldQ, 0, lane);
  v16bf qf1 = frag_a_row(qbase, ldQ, 32, lane);
  v16bf qf2 = frag_a_row(qbase, ldQ, 64, lane);

  v8f o0 = v8f_zero(), o1 = v8f_zero(), o2 = v8f_zero(), o3 = v8f_zero(), o4 = v8f_zero();
  if (lane < 16) { mrow[lane] = -3.0e38f; lrow[lane] = 0.0f; }
  __syncthreads();

  const int ldK = NKV_ * HDP_;  // 768
  const int nkb = (qt * 16 + 15) / 32 + 1;

  for (int kb = 0; kb < nkb; ++kb) {
    const int key0 = kb * 32;
    // ---- S = Q * K^T  (two 16-key n-tiles, 3 K-steps over padded head dim)
    const bf16_t* kb0 = Kp + ((size_t)(b * S_ + key0)) * ldK + (size_t)kvh * HDP_;
    const bf16_t* kb1 = kb0 + (size_t)16 * ldK;
    if (kb + 1 < nkb) __builtin_prefetch(kb0 + (size_t)32 * ldK, 0, 1);
    v8f s0 = v8f_zero(), s1 = v8f_zero();
#pragma unroll
    for (int kk = 0; kk < 3; ++kk) {
      v16bf bt0 = frag_b_rowT(kb0, ldK, kk * 32, lane);
      v16bf bt1 = frag_b_rowT(kb1, ldK, kk * 32, lane);
      v16bf qa = (kk == 0) ? qf0 : (kk == 1) ? qf1 : qf2;
      s0 = wmma_bf16(qa, bt0, s0);
      s1 = wmma_bf16(qa, bt1, s1);
    }
    // ---- spill S tile to LDS
#pragma unroll
    for (int i = 0; i < 8; ++i) {
      Sls[i + 8 * half][n]      = s0[i];
      Sls[i + 8 * half][16 + n] = s1[i];
    }
    // ---- stage V^T into LDS (lane = local key)
    {
      const bf16_t* vrow = Vr + ((size_t)(b * S_ + key0 + lane)) * KVCOLS + (size_t)kvh * HD_;
#pragma unroll
      for (int h8 = 0; h8 < 80; h8 += 8) {
        bf16_t tmp[8];
        *(uint4*)tmp = *(const uint4*)(vrow + h8);
#pragma unroll
        for (int j = 0; j < 8; ++j) Vls[h8 + j][lane] = tmp[j];
      }
    }
    __syncthreads();

    // ---- online softmax (fp32), lanes 0..15 each own one q row
    if (lane < 16) {
      const int r = lane;
      const int qpos = qt * 16 + r;
      float mold = mrow[r];
      float mx = mold;
      float vv[32];
#pragma unroll
      for (int c = 0; c < 32; ++c) {
        float v = Sls[r][c] * INV_NORM_;
        if (key0 + c > qpos) v = NEG_;
        vv[c] = v;
        mx = fmaxf(mx, v);
      }
      float corr = __expf(mold - mx);
      float l = lrow[r] * corr;
#pragma unroll
      for (int c = 0; c < 32; ++c) {
        float p = __expf(vv[c] - mx);
        l += p;
        Pls[r][c] = (bf16_t)p;
      }
      mrow[r] = mx; lrow[r] = l; crow[r] = corr;
    }
    __syncthreads();

    // ---- rescale accumulator rows, then O += P * V
    float cr[8];
#pragma unroll
    for (int i = 0; i < 8; ++i) cr[i] = crow[i + 8 * half];
#pragma unroll
    for (int i = 0; i < 8; ++i) {
      o0[i] *= cr[i]; o1[i] *= cr[i]; o2[i] *= cr[i]; o3[i] *= cr[i]; o4[i] *= cr[i];
    }
    v16bf pa = frag_a_row(&Pls[0][0], 32, 0, lane);
    o0 = wmma_bf16(pa, frag_b_rowT(&Vls[0][0],  32, 0, lane), o0);
    o1 = wmma_bf16(pa, frag_b_rowT(&Vls[16][0], 32, 0, lane), o1);
    o2 = wmma_bf16(pa, frag_b_rowT(&Vls[32][0], 32, 0, lane), o2);
    o3 = wmma_bf16(pa, frag_b_rowT(&Vls[48][0], 32, 0, lane), o3);
    o4 = wmma_bf16(pa, frag_b_rowT(&Vls[64][0], 32, 0, lane), o4);
    __syncthreads();
  }

  // ---- finalize: divide by row sums, write [NT, NQ*80] bf16
  float li[8];
#pragma unroll
  for (int i = 0; i < 8; ++i) li[i] = 1.0f / lrow[i + 8 * half];
  bf16_t* obase = attn_out + qrow0 * QCOLS + (size_t)u * HD_;
#pragma unroll
  for (int i = 0; i < 8; ++i) {
    bf16_t* orow = obase + (size_t)(i + 8 * half) * QCOLS;
    orow[0 * 16 + n] = (bf16_t)(o0[i] * li[i]);
    orow[1 * 16 + n] = (bf16_t)(o1[i] * li[i]);
    orow[2 * 16 + n] = (bf16_t)(o2[i] * li[i]);
    orow[3 * 16 + n] = (bf16_t)(o3[i] * li[i]);
    orow[4 * 16 + n] = (bf16_t)(o4[i] * li[i]);
  }
}

// ---------------------------------------------------------------- launch
extern "C" void kernel_launch(void* const* d_in, const int* in_sizes, int n_in,
                              void* d_out, int out_size, void* d_ws, size_t ws_size,
                              hipStream_t stream) {
  const float* hs = (const float*)d_in[0];   // [B,S,M]
  // d_in[1] = attention_mask (pure causal; recomputed in-kernel)
  const float* wq = (const float*)d_in[2];   // [M, NQ*HD]
  const float* wk = (const float*)d_in[3];   // [M, NKV*HD]
  const float* wv = (const float*)d_in[4];   // [M, NKV*HD]
  const float* wo = (const float*)d_in[5];   // [NQ*HD, M]
  float* out = (float*)d_out;                // [B,S,M] f32
  (void)in_sizes; (void)n_in; (void)out_size; (void)ws_size;

  char* ws = (char*)d_ws;
  size_t off = 0;
  auto alloc = [&](size_t bytes) -> void* {
    void* p = ws + off;
    off += (bytes + 255) & ~(size_t)255;
    return p;
  };

  bf16_t* Xbf   = (bf16_t*)alloc((size_t)NT_ * M_ * 2);
  bf16_t* WqT   = (bf16_t*)alloc((size_t)QCOLS * M_ * 2);
  bf16_t* WkT   = (bf16_t*)alloc((size_t)KVCOLS * M_ * 2);
  bf16_t* WvT   = (bf16_t*)alloc((size_t)KVCOLS * M_ * 2);
  bf16_t* WoT   = (bf16_t*)alloc((size_t)M_ * QCOLS * 2);
  bf16_t* qraw  = (bf16_t*)alloc((size_t)NT_ * QCOLS * 2);
  bf16_t* kraw  = (bf16_t*)alloc((size_t)NT_ * KVCOLS * 2);
  bf16_t* vraw  = (bf16_t*)alloc((size_t)NT_ * KVCOLS * 2);
  bf16_t* Qp    = (bf16_t*)alloc((size_t)NT_ * NQ_ * HDP_ * 2);
  bf16_t* Kp    = (bf16_t*)alloc((size_t)NT_ * NKV_ * HDP_ * 2);
  bf16_t* attnb = (bf16_t*)alloc((size_t)NT_ * QCOLS * 2);

  const int TPB = 256;
  // 1) fp32 -> bf16 conversions (+ weight transposes for B-fragment locality)
  {
    int n = NT_ * M_;
    k_convert<<<(n + TPB - 1) / TPB, TPB, 0, stream>>>(hs, Xbf, n);
  }
  {
    int n = M_ * QCOLS;
    k_convert_t<<<(n + TPB - 1) / TPB, TPB, 0, stream>>>(wq, WqT, M_, QCOLS);
  }
  {
    int n = M_ * KVCOLS;
    k_convert_t<<<(n + TPB - 1) / TPB, TPB, 0, stream>>>(wk, WkT, M_, KVCOLS);
    k_convert_t<<<(n + TPB - 1) / TPB, TPB, 0, stream>>>(wv, WvT, M_, KVCOLS);
  }
  {
    int n = QCOLS * M_;
    k_convert_t<<<(n + TPB - 1) / TPB, TPB, 0, stream>>>(wo, WoT, QCOLS, M_);
  }

  // 2) QKV projections  (C[NT, N] = X[NT, M] * W)
  k_gemm<false><<<dim3(NT_ / 128, QCOLS / 128), TPB, 0, stream>>>(
      Xbf, WqT, (void*)qraw, M_, M_, M_, QCOLS);
  k_gemm<false><<<dim3(NT_ / 128, KVCOLS / 128), TPB, 0, stream>>>(
      Xbf, WkT, (void*)kraw, M_, M_, M_, KVCOLS);
  k_gemm<false><<<dim3(NT_ / 128, KVCOLS / 128), TPB, 0, stream>>>(
      Xbf, WvT, (void*)vraw, M_, M_, M_, KVCOLS);

  // 3) partial RoPE + pad head dim 80 -> 96
  {
    int n = NT_ * NQ_ * HDP_;
    k_rope_pack<<<(n + TPB - 1) / TPB, TPB, 0, stream>>>(qraw, Qp, NQ_, n);
  }
  {
    int n = NT_ * NKV_ * HDP_;
    k_rope_pack<<<(n + TPB - 1) / TPB, TPB, 0, stream>>>(kraw, Kp, NKV_, n);
  }

  // 4) causal flash attention (GQA 4:1), bf16 out
  k_attn<<<B_ * NQ_ * (S_ / 16), 32, 0, stream>>>(Qp, Kp, vraw, attnb);

  // 5) output projection -> f32 d_out
  k_gemm<true><<<dim3(NT_ / 128, M_ / 128), TPB, 0, stream>>>(
      attnb, WoT, (void*)out, QCOLS, QCOLS, QCOLS, M_);
}
